// Attention_20126216749798
// MI455X (gfx1250) — compile-verified, matrix operand order checked by reading, not run on hardware
//
#include <hip/hip_runtime.h>
#include <hip/hip_bf16.h>

typedef unsigned short u16;
typedef unsigned int   u32;
typedef __attribute__((ext_vector_type(16))) __bf16 v16bf;
typedef __attribute__((ext_vector_type(8)))  float  v8f;
typedef __attribute__((ext_vector_type(8)))  u32    u32x8;

#define NUM_B   64
#define NUM_H   12
#define NTOK    197
#define NP      208      // padded token count for S tiles (13*16)
#define NPK     224      // padded K for PV (7*32)
#define HD      64
#define DIM     768
#define MTOK    (NUM_B * NTOK)        // 12608 (= 788*16)
#define SCALE   0.125f

#define QS_ELEMS ((size_t)NUM_B * NUM_H * NP * HD)    // 10,223,616
#define VT_ELEMS ((size_t)NUM_B * NUM_H * HD * NPK)   // 11,010,048

static __device__ __forceinline__ u16 f2bf(float x) {
  u32 u = __float_as_uint(x);
  u32 r = u + 0x7FFFu + ((u >> 16) & 1u);
  return (u16)(r >> 16);
}
static __device__ __forceinline__ u32 pack2(float lo, float hi) {
  return (u32)f2bf(lo) | ((u32)f2bf(hi) << 16);
}
// Load one 16-bit-element WMMA fragment: two 16B chunks 32B apart.
static __device__ __forceinline__ v16bf ldfrag(const u16* p) {
  uint4 a = *(const uint4*)p;
  uint4 b = *(const uint4*)(p + 16);
  u32x8 t;
  t[0]=a.x; t[1]=a.y; t[2]=a.z; t[3]=a.w;
  t[4]=b.x; t[5]=b.y; t[6]=b.z; t[7]=b.w;
  return __builtin_bit_cast(v16bf, t);
}
static __device__ __forceinline__ v8f wmma_bf16(v16bf a, v16bf b, v8f c) {
  return __builtin_amdgcn_wmma_f32_16x16x32_bf16(false, a, false, b, (short)0, c, false, false);
}

// ---------------------------------------------------------------- zero ws
__global__ __launch_bounds__(256) void zero_kernel(uint4* p, long n16) {
  long i = (long)blockIdx.x * blockDim.x + threadIdx.x;
  if (i < n16) p[i] = make_uint4(0u, 0u, 0u, 0u);
}

// ---------------------------------------------------------------- GEMM
// C[M][N] = A[M][K] * W[N][K]^T (+bias).  BM=BN=128, BK=32, 256 thr = 8 waves.
// EPI==0: QKV epilogue -> scatter bf16 q(scaled)/k/vT.  EPI==1: f32 out+bias.
template <bool A_BF16, int EPI>
__global__ __launch_bounds__(256) void gemm_kernel(
    const void* __restrict__ A, const float* __restrict__ W,
    int M, int K, int N,
    const float* __restrict__ bias0, const float* __restrict__ bias1,
    u16* __restrict__ qs, u16* __restrict__ kk, u16* __restrict__ vT,
    float* __restrict__ outp) {
  __shared__ __align__(16) u16 As[128 * 40];
  __shared__ __align__(16) u16 Bs[128 * 40];

  const int tid  = threadIdx.x;
  const int wave = tid >> 5, lane = tid & 31;
  const int half = lane >> 4, l15 = lane & 15;
  const int koff = half * 8;
  const int wm = wave >> 1, wn = wave & 1;
  const int mBase = blockIdx.y * 128, nBase = blockIdx.x * 128;

  const int lr = tid >> 1;           // 0..127
  const int lc = (tid & 1) * 16;     // 0 or 16
  const int gm = mBase + lr;
  const int gnW = nBase + lr;        // weight row (always in-bounds by grid)

  v8f acc[2][4];
#pragma unroll
  for (int i = 0; i < 2; ++i)
#pragma unroll
    for (int j = 0; j < 4; ++j) acc[i][j] = {};

  const int KT = K / 32;
  for (int kt = 0; kt < KT; ++kt) {
    __syncthreads();
    const int gk = kt * 32 + lc;
    uint4 s0, s1;
    if (A_BF16) {
      const u16* Ab = (const u16*)A;
      if (gm < M) {
        s0 = *(const uint4*)(Ab + (size_t)gm * K + gk);
        s1 = *(const uint4*)(Ab + (size_t)gm * K + gk + 8);
      } else { s0 = make_uint4(0,0,0,0); s1 = s0; }
    } else {
      const float* Af = (const float*)A;
      if (gm < M) {
        const float4* fp = (const float4*)(Af + (size_t)gm * K + gk);
        float4 a = fp[0], b = fp[1], c = fp[2], d = fp[3];
        s0 = make_uint4(pack2(a.x,a.y), pack2(a.z,a.w), pack2(b.x,b.y), pack2(b.z,b.w));
        s1 = make_uint4(pack2(c.x,c.y), pack2(c.z,c.w), pack2(d.x,d.y), pack2(d.z,d.w));
      } else { s0 = make_uint4(0,0,0,0); s1 = s0; }
    }
    *(uint4*)&As[lr * 40 + lc]     = s0;
    *(uint4*)&As[lr * 40 + lc + 8] = s1;

    {
      const float4* fp = (const float4*)(W + (size_t)gnW * K + gk);
      float4 a = fp[0], b = fp[1], c = fp[2], d = fp[3];
      *(uint4*)&Bs[lr * 40 + lc]     = make_uint4(pack2(a.x,a.y), pack2(a.z,a.w), pack2(b.x,b.y), pack2(b.z,b.w));
      *(uint4*)&Bs[lr * 40 + lc + 8] = make_uint4(pack2(c.x,c.y), pack2(c.z,c.w), pack2(d.x,d.y), pack2(d.z,d.w));
    }
    __syncthreads();

    v16bf af[2], bf[4];
#pragma unroll
    for (int mi = 0; mi < 2; ++mi)
      af[mi] = ldfrag(&As[(wm * 32 + mi * 16 + l15) * 40 + koff]);
#pragma unroll
    for (int ni = 0; ni < 4; ++ni)
      bf[ni] = ldfrag(&Bs[(wn * 64 + ni * 16 + l15) * 40 + koff]);
#pragma unroll
    for (int mi = 0; mi < 2; ++mi)
#pragma unroll
      for (int ni = 0; ni < 4; ++ni)
        acc[mi][ni] = wmma_bf16(af[mi], bf[ni], acc[mi][ni]);
  }

  // ---------------- epilogue (C element: M = r + 8*half, N = l15)
#pragma unroll
  for (int mi = 0; mi < 2; ++mi)
#pragma unroll
    for (int ni = 0; ni < 4; ++ni)
#pragma unroll
      for (int r = 0; r < 8; ++r) {
        int om = mBase + wm * 32 + mi * 16 + r + half * 8;
        int on = nBase + wn * 64 + ni * 16 + l15;
        float v = acc[mi][ni][r];
        if (om < M) {
          if (EPI == 0) {
            int b = om / NTOK, n = om % NTOK;
            float bb = (on < DIM) ? bias0[on] : ((on < 2 * DIM) ? 0.f : bias1[on - 2 * DIM]);
            v += bb;
            if (on < DIM) {            // q: pre-scale
              int h = on >> 6, d = on & 63;
              qs[(((size_t)b * NUM_H + h) * NP + n) * HD + d] = f2bf(v * SCALE);
            } else if (on < 2 * DIM) { // k
              int c = on - DIM; int h = c >> 6, d = c & 63;
              kk[(((size_t)b * NUM_H + h) * NP + n) * HD + d] = f2bf(v);
            } else {                   // v transposed
              int c = on - 2 * DIM; int h = c >> 6, d = c & 63;
              vT[(((size_t)b * NUM_H + h) * HD + d) * NPK + n] = f2bf(v);
            }
          } else {
            outp[(size_t)om * N + on] = v + bias0[on];
          }
        }
      }
}

// ---------------------------------------------------------------- attention
// One wave per (b, h, 16-row tile). S=q*k^T (13 WMMA tiles), +rel-pos bias,
// softmax via half-wave shuffles, P->LDS (bf16, A layout), out = P*v.
__global__ __launch_bounds__(32) void attn_kernel(
    const u16* __restrict__ qs, const u16* __restrict__ kk,
    const u16* __restrict__ vT, const float* __restrict__ rpb,
    u16* __restrict__ ao) {
  __shared__ __align__(16) u16 Pl[16 * NPK];

  const int blk = blockIdx.x;
  const int mt = blk % 13;
  const int h  = (blk / 13) % NUM_H;
  const int b  = blk / (13 * NUM_H);
  const int lane = threadIdx.x;
  const int half = lane >> 4, l15 = lane & 15;
  const int koff = half * 8;

  const u16* qb = qs + ((size_t)b * NUM_H + h) * NP * HD;
  const u16* kb = kk + ((size_t)b * NUM_H + h) * NP * HD;
  const u16* vb = vT + ((size_t)b * NUM_H + h) * HD * NPK;

  // zero LDS pad columns 208..223
  {
    u32* pz = (u32*)Pl;
    int id = lane;
#pragma unroll
    for (int k = 0; k < 4; ++k, id += 32)
      pz[(id >> 3) * 112 + 104 + (id & 7)] = 0u;
  }

  // q fragments (K = 64 -> two 32-chunks)
  v16bf qf0 = ldfrag(qb + (size_t)(mt * 16 + l15) * HD + 0  + koff);
  v16bf qf1 = ldfrag(qb + (size_t)(mt * 16 + l15) * HD + 32 + koff);

  v8f accS[13];
#pragma unroll
  for (int j = 0; j < 13; ++j) {
    v16bf k0 = ldfrag(kb + (size_t)(j * 16 + l15) * HD + 0  + koff);
    v16bf k1 = ldfrag(kb + (size_t)(j * 16 + l15) * HD + 32 + koff);
    v8f s = {};
    s = wmma_bf16(qf0, k0, s);
    s = wmma_bf16(qf1, k1, s);
    accS[j] = s;
  }

  // bias + mask + softmax; one matrix row per 16-lane half
#pragma unroll
  for (int r = 0; r < 8; ++r) {
    const int qrow = mt * 16 + r + half * 8;
    float mx = -3.0e38f;
#pragma unroll
    for (int j = 0; j < 13; ++j) {
      const int kcol = j * 16 + l15;
      float val;
      if (kcol >= NTOK) val = -1.0e30f;
      else {
        float bb = 0.f;
        if (qrow < NTOK) {
          int idx;
          if (kcol == 0)      idx = (qrow == 0) ? 731 : 730;
          else if (qrow == 0) idx = 729;
          else {
            int i = qrow - 1, jj = kcol - 1;
            int dy = i / 14 - jj / 14 + 13;
            int dx = i % 14 - jj % 14 + 13;
            idx = dy * 27 + dx;
          }
          bb = rpb[idx * NUM_H + h];
        }
        val = accS[j][r] + bb;
      }
      accS[j][r] = val;
      mx = fmaxf(mx, val);
    }
    mx = fmaxf(mx, __shfl_xor(mx, 8, 32));
    mx = fmaxf(mx, __shfl_xor(mx, 4, 32));
    mx = fmaxf(mx, __shfl_xor(mx, 2, 32));
    mx = fmaxf(mx, __shfl_xor(mx, 1, 32));
    float sum = 0.f;
#pragma unroll
    for (int j = 0; j < 13; ++j) {
      float p = __expf(accS[j][r] - mx);
      accS[j][r] = p;
      sum += p;
    }
    sum += __shfl_xor(sum, 8, 32);
    sum += __shfl_xor(sum, 4, 32);
    sum += __shfl_xor(sum, 2, 32);
    sum += __shfl_xor(sum, 1, 32);
    float inv = 1.f / sum;
#pragma unroll
    for (int j = 0; j < 13; ++j)
      Pl[(r + half * 8) * NPK + j * 16 + l15] = f2bf(accS[j][r] * inv);
  }
  __syncthreads();

  // P fragments (A layout, K = 224 -> 7 chunks)
  v16bf pf[7];
#pragma unroll
  for (int s = 0; s < 7; ++s)
    pf[s] = ldfrag(&Pl[l15 * NPK + s * 32 + koff]);

#pragma unroll
  for (int t = 0; t < 4; ++t) {
    v8f accO = {};
#pragma unroll
    for (int s = 0; s < 7; ++s) {
      v16bf vf = ldfrag(vb + (size_t)(t * 16 + l15) * NPK + s * 32 + koff);
      accO = wmma_bf16(pf[s], vf, accO);
    }
#pragma unroll
    for (int r = 0; r < 8; ++r) {
      int qrow = mt * 16 + r + half * 8;
      if (qrow < NTOK)
        ao[((size_t)b * NTOK + qrow) * DIM + h * HD + t * 16 + l15] = f2bf(accO[r]);
    }
  }
}

// ---------------------------------------------------------------- launcher
extern "C" void kernel_launch(void* const* d_in, const int* in_sizes, int n_in,
                              void* d_out, int out_size, void* d_ws, size_t ws_size,
                              hipStream_t stream) {
  const float* x      = (const float*)d_in[0];
  const float* qkv_w  = (const float*)d_in[1];
  const float* q_bias = (const float*)d_in[2];
  const float* v_bias = (const float*)d_in[3];
  const float* rpb    = (const float*)d_in[4];
  const float* proj_w = (const float*)d_in[5];
  const float* proj_b = (const float*)d_in[6];
  float* out = (float*)d_out;

  u16* qs = (u16*)d_ws;
  u16* kk = qs + QS_ELEMS;
  u16* vT = kk + QS_ELEMS;
  u16* ao = vT + VT_ELEMS;

  // zero q/k/vT (covers WMMA padding)
  const long zbytes = (long)(2 * QS_ELEMS + VT_ELEMS) * 2;
  const long n16 = zbytes / 16;
  zero_kernel<<<dim3((unsigned)((n16 + 255) / 256)), 256, 0, stream>>>((uint4*)d_ws, n16);

  // QKV: [12608,768] x [2304,768]^T
  gemm_kernel<false, 0><<<dim3(2304 / 128, (MTOK + 127) / 128), 256, 0, stream>>>(
      (const void*)x, qkv_w, MTOK, DIM, 3 * DIM, q_bias, v_bias, qs, kk, vT, nullptr);

  // attention: B*H*13 waves
  attn_kernel<<<dim3(NUM_B * NUM_H * 13), 32, 0, stream>>>(qs, kk, vT, rpb, ao);

  // proj: [12608,768] x [768,768]^T + bias -> f32 out
  gemm_kernel<true, 1><<<dim3(768 / 128, (MTOK + 127) / 128), 256, 0, stream>>>(
      (const void*)ao, proj_w, MTOK, DIM, DIM, proj_b, nullptr, nullptr, nullptr, nullptr, out);
}
